// VisionLanguageTransformer_65008624992531
// MI455X (gfx1250) — matrix operand order repaired and checked by reading, not compiled, not run on hardware
//
#include <hip/hip_runtime.h>

// ---------------------------------------------------------------------------
// VisionLanguage transformer forward for gfx1250 (MI455X).
// Compute-bound (~700 GFLOP vs ~0.5 GB weights) => bf16 WMMA with f32 accum.
// ---------------------------------------------------------------------------

#define B_SZ     16
#define T_LEN    256
#define E_DIM    1024
#define H_NUM    16
#define HD_DIM   64
#define FFN_DIM  4096
#define VOCAB_SZ 30000
#define IMG_DIMS 2048
#define TXT_DIMS 768

typedef __bf16 bf16;
typedef bf16  v16bf __attribute__((ext_vector_type(16)));
typedef bf16  v8bf  __attribute__((ext_vector_type(8)));
typedef float v8f   __attribute__((ext_vector_type(8)));

__device__ __forceinline__ bf16 f2bf(float f) { return (bf16)f; }

__device__ __forceinline__ float gelu_exact(float x) {
  return 0.5f * x * (1.0f + erff(x * 0.70710678118654752440f));
}

// Block tile 128x128x64; 256 threads = 8 wave32 arranged 4(M) x 2(N);
// each wave computes 32x64 via 2x4 fragments of v_wmma_f32_16x16x32_bf16.
constexpr int BMt = 128, BNt = 128, BKt = 64;
constexpr int LDK = BKt + 8;   // 144B rows: keeps every 8-elem group 16B aligned

template <bool TRANS_B>
__global__ __launch_bounds__(256) void gemm_wmma_bf16(
    const float* __restrict__ A, long lda, long sAo, long sAi,
    const float* __restrict__ Bm, long ldb, long sBo, long sBi,
    const float* __restrict__ bias,
    const float* __restrict__ res, long ldr, int resMod,
    float* __restrict__ C, long ldc, long sCo, long sCi,
    int M, int N, int K, int nInner, int act, float scale)
{
  const int bz = blockIdx.z;
  const int bo = bz / nInner, bi = bz % nInner;
  A  += (long)bo * sAo + (long)bi * sAi;
  Bm += (long)bo * sBo + (long)bi * sBi;
  C  += (long)bo * sCo + (long)bi * sCi;

  const int tid  = threadIdx.x;
  const int lane = tid & 31;
  const int wave = tid >> 5;
  const int wm   = wave >> 1;   // 4 waves along M (32 rows each)
  const int wn   = wave & 1;    // 2 waves along N (64 cols each)
  const int half = lane >> 4;
  const int l16  = lane & 15;
  const int tM = blockIdx.y * BMt;
  const int tN = blockIdx.x * BNt;
  const int Mm1 = M - 1, Nm1 = N - 1;

  __shared__ bf16 As[BMt][LDK];
  __shared__ bf16 Bs[BNt][LDK];

  v8f acc[2][4] = {};

  // K is a multiple of BKt in every launch from this file; M/N handled by
  // clamped (always-valid) staging addresses + masked epilogue stores.
  for (int k0 = 0; k0 < K; k0 += BKt) {
    // --- prefetch next k-tile (2 cachelines per thread) ---
    if (k0 + BKt < K) {
      {
        int r = tid >> 1, off = (tid & 1) << 5;
        int gr = tM + r; if (gr > Mm1) gr = Mm1;
        __builtin_prefetch(A + (long)gr * lda + (k0 + BKt + off), 0, 3);
      }
      if (TRANS_B) {
        int n = tid >> 1, off = (tid & 1) << 5;
        int gn = tN + n; if (gn > Nm1) gn = Nm1;
        __builtin_prefetch(Bm + (long)gn * ldb + (k0 + BKt + off), 0, 3);
      } else {
        int kr = tid >> 2, off = (tid & 3) << 5;
        __builtin_prefetch(Bm + (long)(k0 + BKt + kr) * ldb + (tN + off), 0, 3);
      }
    }
    // --- stage A tile 128x64: 8 elems/thread/iter, 4 iters ---
#pragma unroll
    for (int i = 0; i < 4; ++i) {
      int s = tid + i * 256;
      int r = s >> 3, ko = (s & 7) << 3;
      int gr = tM + r; if (gr > Mm1) gr = Mm1;
      const float* ap = A + (long)gr * lda + (k0 + ko);
      float4 v0 = *(const float4*)ap;
      float4 v1 = *(const float4*)(ap + 4);
      v8bf p;
      p[0] = f2bf(v0.x); p[1] = f2bf(v0.y); p[2] = f2bf(v0.z); p[3] = f2bf(v0.w);
      p[4] = f2bf(v1.x); p[5] = f2bf(v1.y); p[6] = f2bf(v1.z); p[7] = f2bf(v1.w);
      *(v8bf*)&As[r][ko] = p;
    }
    // --- stage B tile as [N][K] 128x64 ---
#pragma unroll
    for (int i = 0; i < 4; ++i) {
      int s = tid + i * 256;
      int n = s >> 3, ko = (s & 7) << 3;
      int gn = tN + n; if (gn > Nm1) gn = Nm1;
      v8bf p;
      if (TRANS_B) {
        const float* bp = Bm + (long)gn * ldb + (k0 + ko);
        float4 v0 = *(const float4*)bp;
        float4 v1 = *(const float4*)(bp + 4);
        p[0] = f2bf(v0.x); p[1] = f2bf(v0.y); p[2] = f2bf(v0.z); p[3] = f2bf(v0.w);
        p[4] = f2bf(v1.x); p[5] = f2bf(v1.y); p[6] = f2bf(v1.z); p[7] = f2bf(v1.w);
      } else {
        const float* bp = Bm + (long)(k0 + ko) * ldb + gn;
#pragma unroll
        for (int e = 0; e < 8; ++e) p[e] = f2bf(bp[(long)e * ldb]);
      }
      *(v8bf*)&Bs[n][ko] = p;
    }
    __syncthreads();

#pragma unroll
    for (int kk0 = 0; kk0 < BKt; kk0 += 32) {
      v16bf af[2], bfr[4];
#pragma unroll
      for (int i = 0; i < 2; ++i) {
        int row = wm * 32 + i * 16 + l16;
        // A 16x32 striping: lane half picks k runs [half*8, half*8+8) and +16
        v8bf lo = *(const v8bf*)&As[row][kk0 + half * 8];
        v8bf hi = *(const v8bf*)&As[row][kk0 + 16 + half * 8];
#pragma unroll
        for (int e = 0; e < 8; ++e) { af[i][e] = lo[e]; af[i][8 + e] = hi[e]; }
      }
#pragma unroll
      for (int j = 0; j < 4; ++j) {
        int col = wn * 64 + j * 16 + l16;
        // B 32x16 striping: lane half picks k runs [half*8, half*8+8) and +16
        v8bf b0 = *(const v8bf*)&Bs[col][kk0 + half * 8];
        v8bf b1 = *(const v8bf*)&Bs[col][kk0 + 16 + half * 8];
#pragma unroll
        for (int e = 0; e < 8; ++e) { bfr[j][e] = b0[e]; bfr[j][8 + e] = b1[e]; }
      }
#pragma unroll
      for (int i = 0; i < 2; ++i)
#pragma unroll
        for (int j = 0; j < 4; ++j)
          acc[i][j] = __builtin_amdgcn_wmma_f32_16x16x32_bf16(
              false, af[i], false, bfr[j], (short)0, acc[i][j], false, false);
    }
    __syncthreads();
  }

  // --- epilogue: scale, bias, residual(+mod), activation, masked stores ---
#pragma unroll
  for (int i = 0; i < 2; ++i) {
#pragma unroll
    for (int j = 0; j < 4; ++j) {
      int col = tN + wn * 64 + j * 16 + l16;
      if (col >= N) continue;
#pragma unroll
      for (int r = 0; r < 8; ++r) {
        int row = tM + wm * 32 + i * 16 + half * 8 + r;
        if (row >= M) continue;
        float v = acc[i][j][r] * scale;
        if (bias) v += bias[col];
        if (res) {
          long rr = resMod ? (long)(row % resMod) : (long)row;
          v += res[rr * ldr + col];
        }
        if (act == 1) v = gelu_exact(v);
        C[(long)row * ldc + col] = v;
      }
    }
  }
}

// out[row] = g * (x+h - mu) / sqrt(var+eps) + b    (row length = E_DIM)
__global__ __launch_bounds__(256) void ln_fused(
    const float* __restrict__ x, const float* __restrict__ h,
    const float* __restrict__ g, const float* __restrict__ b,
    float* __restrict__ out)
{
  const int D = E_DIM;
  long row = blockIdx.x;
  const float* xr = x + row * D;
  const float* hr = h ? h + row * D : nullptr;
  float vals[4];
  float s = 0.f;
#pragma unroll
  for (int i = 0; i < 4; ++i) {
    int c = threadIdx.x + i * 256;
    float v = xr[c] + (hr ? hr[c] : 0.f);
    vals[i] = v; s += v;
  }
  __shared__ float red[256];
  red[threadIdx.x] = s; __syncthreads();
  for (int st = 128; st > 0; st >>= 1) {
    if ((int)threadIdx.x < st) red[threadIdx.x] += red[threadIdx.x + st];
    __syncthreads();
  }
  float mu = red[0] * (1.f / D); __syncthreads();
  float vs = 0.f;
#pragma unroll
  for (int i = 0; i < 4; ++i) { float d = vals[i] - mu; vs += d * d; }
  red[threadIdx.x] = vs; __syncthreads();
  for (int st = 128; st > 0; st >>= 1) {
    if ((int)threadIdx.x < st) red[threadIdx.x] += red[threadIdx.x + st];
    __syncthreads();
  }
  float inv = rsqrtf(red[0] * (1.f / D) + 1e-5f);
#pragma unroll
  for (int i = 0; i < 4; ++i) {
    int c = threadIdx.x + i * 256;
    out[row * D + c] = g[c] * (vals[i] - mu) * inv + b[c];
  }
}

__global__ __launch_bounds__(256) void softmax_rows(float* __restrict__ s, int L)
{
  long row = blockIdx.x;
  float* r = s + row * (long)L;
  int t = threadIdx.x;
  __shared__ float red[256];
  float m = -3.0e38f;
  for (int c = t; c < L; c += 256) m = fmaxf(m, r[c]);
  red[t] = m; __syncthreads();
  for (int st = 128; st > 0; st >>= 1) {
    if (t < st) red[t] = fmaxf(red[t], red[t + st]);
    __syncthreads();
  }
  m = red[0]; __syncthreads();
  float s0 = 0.f;
  for (int c = t; c < L; c += 256) { float e = expf(r[c] - m); r[c] = e; s0 += e; }
  red[t] = s0; __syncthreads();
  for (int st = 128; st > 0; st >>= 1) {
    if (t < st) red[t] += red[t + st];
    __syncthreads();
  }
  float inv = 1.f / red[0];
  for (int c = t; c < L; c += 256) r[c] *= inv;
}

// Lk==1 attention: softmax of a single key is 1 => ctx[b,q,:] = V[b,0,:]
__global__ void bcast_ctx(const float* __restrict__ V, float* __restrict__ ctx,
                          int Lq, long total)
{
  long i = (long)blockIdx.x * 256 + threadIdx.x;
  if (i >= total) return;
  long e = i % E_DIM;
  long b = (i / E_DIM) / Lq;
  ctx[i] = V[b * E_DIM + e];
}

__global__ void gather_row0(const float* __restrict__ src, float* __restrict__ dst,
                            long srcRowStride, long total)
{
  long i = (long)blockIdx.x * 256 + threadIdx.x;
  if (i >= total) return;
  long b = i / E_DIM, e = i % E_DIM;
  dst[i] = src[b * srcRowStride + e];
}

__global__ __launch_bounds__(256) void l2norm_rows(float* __restrict__ x)
{
  const int D = E_DIM;
  long row = blockIdx.x;
  float* r = x + row * D;
  float s = 0.f;
#pragma unroll
  for (int i = 0; i < 4; ++i) { float v = r[threadIdx.x + i * 256]; s += v * v; }
  __shared__ float red[256];
  red[threadIdx.x] = s; __syncthreads();
  for (int st = 128; st > 0; st >>= 1) {
    if ((int)threadIdx.x < st) red[threadIdx.x] += red[threadIdx.x + st];
    __syncthreads();
  }
  float inv = 1.f / sqrtf(red[0]);
#pragma unroll
  for (int i = 0; i < 4; ++i) r[threadIdx.x + i * 256] *= inv;
}

__global__ void itc_kernel(const float* __restrict__ vi, const float* __restrict__ ti,
                           const float* __restrict__ ls, float* __restrict__ out)
{
  int i = threadIdx.x >> 4, j = threadIdx.x & 15;   // 16x16
  float s = 0.f;
  for (int d = 0; d < E_DIM; ++d) s += vi[i * E_DIM + d] * ti[j * E_DIM + d];
  out[i * B_SZ + j] = expf(ls[0]) * s;
}

__global__ void concat_pool(const float* __restrict__ vp, const float* __restrict__ tp,
                            float* __restrict__ cat, long total)
{
  long i = (long)blockIdx.x * 256 + threadIdx.x;
  if (i >= total) return;
  long b = i / (2 * E_DIM), c = i % (2 * E_DIM);
  cat[i] = (c < E_DIM) ? vp[b * E_DIM + c] : tp[b * E_DIM + (c - E_DIM)];
}

// Block param leaf order (jax tree flatten, sorted dict keys):
// f1.b f1.w f2.b f2.w k.b k.w ln1.b ln1.g ln2.b ln2.g o.b o.w q.b q.w v.b v.w
enum {
  LF_F1B = 0, LF_F1W, LF_F2B, LF_F2W, LF_KB, LF_KW, LF_LN1B, LF_LN1G,
  LF_LN2B, LF_LN2G, LF_OB, LF_OW, LF_QB, LF_QW, LF_VB, LF_VW
};

extern "C" void kernel_launch(void* const* d_in, const int* in_sizes, int n_in,
                              void* d_out, int out_size, void* d_ws, size_t ws_size,
                              hipStream_t stream)
{
  (void)in_sizes; (void)out_size; (void)ws_size;
  if (n_in < 150) return;   // expected jax-flattened leaf count

  // Top level: image_features, text_features, then params (sorted keys).
  enum {
    IN_IMG = 0, IN_TXT = 1,
    IMG_PROJ_B = 2, IMG_PROJ_W = 3, ITM1_B = 4, ITM1_W = 5, ITM2_B = 6, ITM2_W = 7,
    LOGIT_SCALE = 8, MLM1_B = 9, MLM1_W = 10, MLM2_B = 11, MLM2_W = 12,
    MLM_LN_B = 13, MLM_LN_G = 14,
    T2V0 = 15, T2V1 = 31, TEXT0 = 47, TEXT1 = 63,
    TEXT_POS = 79, TEXT_PROJ_B = 80, TEXT_PROJ_W = 81, TPROJ_B = 82, TPROJ_W = 83,
    V2T0 = 84, V2T1 = 100, VIS0 = 116, VIS1 = 132, VPROJ_B = 148, VPROJ_W = 149
  };
  auto F = [&](int i) { return (const float*)d_in[i]; };

  // --- workspace carve ---
  float* ws = (float*)d_ws;
  size_t off = 0;
  auto alloc = [&](size_t n) { float* p = ws + off; off += n; return p; };
  const size_t TE = (size_t)B_SZ * T_LEN * E_DIM;
  float* t0  = alloc(TE);
  float* t1  = alloc(TE);
  float* v0  = alloc((size_t)B_SZ * E_DIM);
  float* v1  = alloc((size_t)B_SZ * E_DIM);
  float* Qb  = alloc(TE);
  float* Kb  = alloc(TE);
  float* Vb  = alloc(TE);
  float* Cx  = alloc(TE);
  float* big = alloc((size_t)B_SZ * T_LEN * FFN_DIM);   // scores OR ffn (disjoint use)
  float* tmp = alloc(TE);
  float* tpool = alloc((size_t)B_SZ * E_DIM);
  float* vi    = alloc((size_t)B_SZ * E_DIM);
  float* ti    = alloc((size_t)B_SZ * E_DIM);
  float* cat   = alloc((size_t)B_SZ * 2 * E_DIM);
  float* itmh  = alloc((size_t)B_SZ * E_DIM);

  auto gemm = [&](bool transB,
                  const float* A, long lda, long sAo, long sAi,
                  const float* Bm, long ldb, long sBo, long sBi,
                  const float* bias, const float* res, long ldr, int resMod,
                  float* C, long ldc, long sCo, long sCi,
                  int M, int N, int K, int bo, int bi, int act, float scale) {
    dim3 grid((N + BNt - 1) / BNt, (M + BMt - 1) / BMt, bo * bi);
    if (transB)
      gemm_wmma_bf16<true><<<grid, 256, 0, stream>>>(A, lda, sAo, sAi, Bm, ldb, sBo, sBi,
          bias, res, ldr, resMod, C, ldc, sCo, sCi, M, N, K, bi, act, scale);
    else
      gemm_wmma_bf16<false><<<grid, 256, 0, stream>>>(A, lda, sAo, sAi, Bm, ldb, sBo, sBi,
          bias, res, ldr, resMod, C, ldc, sCo, sCi, M, N, K, bi, act, scale);
  };

  // y[M,N] = act(x[M,K] @ W[N,K]^T + b [+ res])
  auto linT = [&](const float* A, const float* W, const float* bias, float* C,
                  int M, int N, int K, int act,
                  const float* res = nullptr, long ldr = 0, int resMod = 0) {
    gemm(true, A, K, 0, 0, W, K, 0, 0, bias, res, ldr, resMod,
         C, N, 0, 0, M, N, K, 1, 1, act, 1.f);
  };

  // x <- block(x, kv) in place. x:[B*Lq,E], kv:[B*Lk,E]
  auto run_block = [&](float* x, const float* kv, int Lq, int Lk, int base) {
    const float* P[16];
    for (int i = 0; i < 16; ++i) P[i] = F(base + i);
    int Mq = B_SZ * Lq, Mk = B_SZ * Lk;
    linT(x,  P[LF_QW], P[LF_QB], Qb, Mq, E_DIM, E_DIM, 0);
    linT(kv, P[LF_KW], P[LF_KB], Kb, Mk, E_DIM, E_DIM, 0);
    linT(kv, P[LF_VW], P[LF_VB], Vb, Mk, E_DIM, E_DIM, 0);
    if (Lk == 1) {
      long total = (long)Mq * E_DIM;
      bcast_ctx<<<(total + 255) / 256, 256, 0, stream>>>(Vb, Cx, Lq, total);
    } else {
      // scores[b,h,q,k] = (Q . K) / sqrt(HD)
      gemm(true, Qb, E_DIM, (long)Lq * E_DIM, HD_DIM,
                 Kb, E_DIM, (long)Lk * E_DIM, HD_DIM,
                 nullptr, nullptr, 0, 0,
                 big, Lk, (long)H_NUM * Lq * Lk, (long)Lq * Lk,
                 Lq, Lk, HD_DIM, B_SZ, H_NUM, 0, 0.125f);
      softmax_rows<<<B_SZ * H_NUM * Lq, 256, 0, stream>>>(big, Lk);
      // ctx[b,q,h,:] = w @ V
      gemm(false, big, Lk, (long)H_NUM * Lq * Lk, (long)Lq * Lk,
                  Vb, E_DIM, (long)Lk * E_DIM, HD_DIM,
                  nullptr, nullptr, 0, 0,
                  Cx, E_DIM, (long)Lq * E_DIM, HD_DIM,
                  Lq, HD_DIM, Lk, B_SZ, H_NUM, 0, 1.f);
    }
    linT(Cx, P[LF_OW], P[LF_OB], tmp, Mq, E_DIM, E_DIM, 0);
    ln_fused<<<Mq, 256, 0, stream>>>(x, tmp, P[LF_LN1G], P[LF_LN1B], x);
    linT(x, P[LF_F1W], P[LF_F1B], big, Mq, FFN_DIM, E_DIM, 1 /*gelu*/);
    linT(big, P[LF_F2W], P[LF_F2B], tmp, Mq, E_DIM, FFN_DIM, 0);
    ln_fused<<<Mq, 256, 0, stream>>>(x, tmp, P[LF_LN2G], P[LF_LN2B], x);
  };

  // --- projections ---
  linT(F(IN_IMG), F(IMG_PROJ_W), F(IMG_PROJ_B), v0, B_SZ, E_DIM, IMG_DIMS, 0);
  linT(F(IN_TXT), F(TEXT_PROJ_W), F(TEXT_PROJ_B), t0, B_SZ * T_LEN, E_DIM, TXT_DIMS, 0,
       F(TEXT_POS), E_DIM, T_LEN /* pos row = row % T */);

  // --- unimodal encoders ---
  run_block(v0, v0, 1, 1, VIS0);
  run_block(v0, v0, 1, 1, VIS1);
  run_block(t0, t0, T_LEN, T_LEN, TEXT0);
  run_block(t0, t0, T_LEN, T_LEN, TEXT1);

  // --- cross blocks (ping-pong buffers: v2/t2 from old v,t) ---
  float *vc = v0, *vo = v1, *tc = t0, *to = t1;
  const int V2T[2] = {V2T0, V2T1}, T2V[2] = {T2V0, T2V1};
  for (int i = 0; i < 2; ++i) {
    hipMemcpyAsync(vo, vc, sizeof(float) * B_SZ * E_DIM, hipMemcpyDeviceToDevice, stream);
    hipMemcpyAsync(to, tc, sizeof(float) * TE, hipMemcpyDeviceToDevice, stream);
    run_block(vo, tc, 1, T_LEN, V2T[i]);       // v2 = block(v, t, pv)
    run_block(to, vc, T_LEN, 1, T2V[i]);       // t2 = block(t, v, pt)
    float* s;
    s = vc; vc = vo; vo = s;
    s = tc; tc = to; to = s;
  }

  float* out = (float*)d_out;
  // --- pools ---
  {
    long total = (long)B_SZ * E_DIM;
    gather_row0<<<(total + 255) / 256, 256, 0, stream>>>(tc, tpool, (long)T_LEN * E_DIM, total);
  }
  // --- ITC ---
  linT(vc, F(VPROJ_W), F(VPROJ_B), vi, B_SZ, E_DIM, E_DIM, 0);
  l2norm_rows<<<B_SZ, 256, 0, stream>>>(vi);
  linT(tpool, F(TPROJ_W), F(TPROJ_B), ti, B_SZ, E_DIM, E_DIM, 0);
  l2norm_rows<<<B_SZ, 256, 0, stream>>>(ti);
  itc_kernel<<<1, 256, 0, stream>>>(vi, ti, F(LOGIT_SCALE), out);
  // --- ITM ---
  {
    long total = (long)B_SZ * 2 * E_DIM;
    concat_pool<<<(total + 255) / 256, 256, 0, stream>>>(vc, tpool, cat, total);
  }
  linT(cat, F(ITM1_W), F(ITM1_B), itmh, B_SZ, E_DIM, 2 * E_DIM, 1 /*gelu*/);
  linT(itmh, F(ITM2_W), F(ITM2_B), out + 256, B_SZ, 2, E_DIM, 0);
  // --- MLM ---
  linT(tc, F(MLM1_W), F(MLM1_B), tmp, B_SZ * T_LEN, E_DIM, E_DIM, 1 /*gelu*/);
  ln_fused<<<B_SZ * T_LEN, 256, 0, stream>>>(tmp, nullptr, F(MLM_LN_G), F(MLM_LN_B), Qb);
  linT(Qb, F(MLM2_W), F(MLM2_B), out + 256 + 32, B_SZ * T_LEN, VOCAB_SZ, E_DIM, 0);
}
